// FeatureMapSelfAttention_90331752169774
// MI455X (gfx1250) — matrix operand order repaired and checked by reading, not compiled, not run on hardware
//
#include <hip/hip_runtime.h>

// ---------------------------------------------------------------------------
// CDNA5 (gfx1250) SAGAN feature-map self-attention, bf16 WMMA pipeline.
//   B=8, C=256, H=W=64, N=4096
//   q = Wq f + bq ; k = Wk f + bk ; v = Wv f + bv          (bf16 WMMA, f32 acc)
//   sT[m][n] = (1/W) * sum_c q[c,n] k[c,m]                 (bf16 WMMA, f32 acc)
//   attn[:,m] = softmax over n  ->  attnT[m][n] bf16
//   o[c,m] = sum_n v[c,n] attnT[m][n]                      (bf16 WMMA, f32 acc)
//   out = gamma * o + x
// Heavy GEMMs use 2x2 register blocking (32x32 per wave, 4 accumulators).
// QKV kernel stages its weight panel to LDS with async global->LDS loads.
// Batches processed sequentially on `stream` so workspace (~104MB) is reused.
// ---------------------------------------------------------------------------

typedef __bf16 bf16;
typedef __attribute__((ext_vector_type(8)))  bf16  v8bf;
typedef __attribute__((ext_vector_type(16))) bf16  v16bf;
typedef __attribute__((ext_vector_type(8)))  float v8f;
typedef __attribute__((ext_vector_type(4)))  int   v4i;

typedef __attribute__((address_space(1))) v4i* as1_v4i;
typedef __attribute__((address_space(3))) v4i* as3_v4i;

// ---- CDNA5 async global->LDS helpers ---------------------------------------

__device__ inline void async_copy_b128(const void* gsrc, void* lds_dst) {
#if __has_builtin(__builtin_amdgcn_global_load_async_to_lds_b128)
  __builtin_amdgcn_global_load_async_to_lds_b128(
      (as1_v4i)gsrc, (as3_v4i)lds_dst, 0, 0);
#else
  __attribute__((address_space(3))) void* l =
      (__attribute__((address_space(3))) void*)lds_dst;
  unsigned loff = (unsigned)(unsigned long)l;
  asm volatile("global_load_async_to_lds_b128 %0, %1, off"
               :: "v"(loff), "v"((unsigned long long)gsrc) : "memory");
#endif
}

__device__ inline void wait_async0() {
#if __has_builtin(__builtin_amdgcn_s_wait_asynccnt)
  __builtin_amdgcn_s_wait_asynccnt(0);
#else
  asm volatile("s_wait_asynccnt 0" ::: "memory");
#endif
}

// ---- WMMA fragment loaders (contiguous-k layouts, wave32) --------------------

// A-matrix 16x32 bf16, row-major source, row stride ld (elements).
// lanes 0-15 : row M=lane,   K = 0..15
// lanes 16-31: row M=lane-16,K = 16..31
__device__ inline v16bf load_a_frag(const bf16* base, int ld, int lane) {
  const int r  = lane & 15;
  const int hi = (lane >> 4) & 1;
  const bf16* p = base + r * ld + hi * 16;
  v8bf lo = *(const v8bf*)(p);        // K = hi*16 + 0..7
  v8bf hv = *(const v8bf*)(p + 8);    // K = hi*16 + 8..15
  return __builtin_shufflevector(lo, hv, 0,1,2,3,4,5,6,7,8,9,10,11,12,13,14,15);
}

// B-matrix 32x16 bf16 from a column-major source (K contiguous), stride ld.
// lanes 0-15 : col N=lane,    K = 0..15
// lanes 16-31: col N=lane-16, K = 16..31
__device__ inline v16bf load_b_frag(const bf16* base, int ld, int lane) {
  const bf16* p = base + (lane & 15) * ld + ((lane >> 4) & 1) * 16;
  return *(const v16bf*)p;            // 16 contiguous bf16 (32B)
}

__device__ inline v8f wmma_bf16(v16bf a, v16bf b, v8f c) {
  return __builtin_amdgcn_wmma_f32_16x16x32_bf16(false, a, false, b, (short)0, c,
                                                 false, false);
}

// ---- pack kernels ----------------------------------------------------------

__global__ __launch_bounds__(256)
void pack_w_kernel(const float* __restrict__ W, bf16* __restrict__ Wb, int n) {
  int i = blockIdx.x * 256 + threadIdx.x;
  if (i < n) Wb[i] = (bf16)W[i];
}

// xT[n][c] (bf16) from x[c][n] (f32)
__global__ __launch_bounds__(256)
void pack_xT_kernel(const float* __restrict__ x, bf16* __restrict__ xT, int C, int N) {
  int i = blockIdx.x * 256 + threadIdx.x;
  if (i < N * C) {
    int n = i / C, c = i - n * C;
    xT[i] = (bf16)x[(size_t)c * N + n];
  }
}

// ---- QKV projection: D[d][n] = sum_c W[d][c] * xT[n][c] + b[d] -------------
// z = blockIdx.z selects (Wq->qT, Wk->kT, Wv->v).
// Each wave async-stages its 16xC weight panel into LDS, then runs the K loop
// with A-fragments served from LDS (ds_load_b128) and B-fragments from global.
#define LDP 264   // padded LDS row stride in bf16 (264*2 bytes, 16B multiple)

__global__ __launch_bounds__(128)
void qkv_kernel(const bf16* __restrict__ Wqb, const bf16* __restrict__ Wkb,
                const bf16* __restrict__ Wvb,
                const float* __restrict__ bq, const float* __restrict__ bk,
                const float* __restrict__ bv,
                const bf16* __restrict__ xT,
                bf16* __restrict__ qT, bf16* __restrict__ kT, bf16* __restrict__ v,
                int C, int N) {
  __shared__ bf16 sW[4][16 * LDP];

  const int lane = threadIdx.x & 31;
  const int wave = threadIdx.x >> 5;
  const int tile = blockIdx.x * 4 + wave;        // C/16 * N/16 tiles
  const int tdN  = C >> 4;                       // 16 tiles along d
  const int td   = (tile % tdN) * 16;
  const int tn   = (tile / tdN) * 16;
  const int z    = blockIdx.z;

  const bf16*  Wm = (z == 0) ? Wqb : (z == 1) ? Wkb : Wvb;
  const float* bb = (z == 0) ? bq  : (z == 1) ? bk  : bv;

  // async-stage W[td..td+16) x [0..C) -> LDS (wave-private panel)
  {
    const bf16* gpanel = Wm + (size_t)td * C;
    bf16* lpanel = &sW[wave][0];
#pragma unroll
    for (int t = 0; t < 16; ++t) {            // one row (C*2 = 512B) per step
      async_copy_b128(gpanel + t * C + lane * 8, lpanel + t * LDP + lane * 8);
    }
    wait_async0();
  }

  v8f acc = {};
  for (int kt = 0; kt < C; kt += 32) {
    v16bf a = load_a_frag(&sW[wave][kt], LDP, lane);            // LDS (ds_load)
    v16bf b = load_b_frag(xT + (size_t)tn * C + kt, C, lane);   // global
    acc = wmma_bf16(a, b, acc);
  }

  const int n  = tn + (lane & 15);
  const int d0 = td + ((lane >> 4) ? 8 : 0);
  if (z < 2) {
    bf16* dst = ((z == 0) ? qT : kT) + (size_t)n * C + d0;
    v8bf o;
#pragma unroll
    for (int j = 0; j < 8; ++j) o[j] = (bf16)(acc[j] + bb[d0 + j]);
    *(v8bf*)dst = o;
  } else {
#pragma unroll
    for (int j = 0; j < 8; ++j)
      v[(size_t)(d0 + j) * N + n] = (bf16)(acc[j] + bb[d0 + j]);
  }
}

// ---- scores: sT[m][n] = scale * sum_c kT[m][c] * qT[n][c], 2x2 blocked -----

__device__ inline void store_s_tile(float* __restrict__ sT, int N, int tm, int tn,
                                    int lane, v8f acc, float scale) {
  const int n  = tn + (lane & 15);
  const int m0 = tm + ((lane >> 4) ? 8 : 0);
#pragma unroll
  for (int j = 0; j < 8; ++j)
    sT[(size_t)(m0 + j) * N + n] = acc[j] * scale;
}

__global__ __launch_bounds__(128)
void qk_kernel(const bf16* __restrict__ qT, const bf16* __restrict__ kT,
               float* __restrict__ sT, int C, int N, float scale) {
  const int lane = threadIdx.x & 31;
  const int wave = threadIdx.x >> 5;
  const int tile = blockIdx.x * 4 + wave;        // (N/32)^2 super-tiles
  const int tmN  = N >> 5;                       // 128
  const int tm   = (tile % tmN) * 32;
  const int tn   = (tile / tmN) * 32;

  v8f a00 = {}, a01 = {}, a10 = {}, a11 = {};
  for (int kt = 0; kt < C; kt += 32) {
    __builtin_prefetch(kT + (size_t)tm * C + kt + 32, 0, 0);
    v16bf ka0 = load_a_frag(kT + (size_t)(tm     ) * C + kt, C, lane);
    v16bf ka1 = load_a_frag(kT + (size_t)(tm + 16) * C + kt, C, lane);
    v16bf qb0 = load_b_frag(qT + (size_t)(tn     ) * C + kt, C, lane);
    v16bf qb1 = load_b_frag(qT + (size_t)(tn + 16) * C + kt, C, lane);
    a00 = wmma_bf16(ka0, qb0, a00);
    a01 = wmma_bf16(ka0, qb1, a01);
    a10 = wmma_bf16(ka1, qb0, a10);
    a11 = wmma_bf16(ka1, qb1, a11);
  }
  store_s_tile(sT, N, tm,      tn,      lane, a00, scale);
  store_s_tile(sT, N, tm,      tn + 16, lane, a01, scale);
  store_s_tile(sT, N, tm + 16, tn,      lane, a10, scale);
  store_s_tile(sT, N, tm + 16, tn + 16, lane, a11, scale);
}

// ---- softmax over n (rows of sT), emit bf16 attnT[m][n] --------------------

__global__ __launch_bounds__(256)
void softmax_kernel(const float* __restrict__ sT, bf16* __restrict__ attnT, int N) {
  const int m = blockIdx.x;
  const float* row = sT + (size_t)m * N;
  bf16* orow = attnT + (size_t)m * N;
  __shared__ float red[256];
  const int tid = threadIdx.x;

  float mx = -__builtin_inff();
  for (int i = tid; i < N; i += 256) mx = fmaxf(mx, row[i]);
  red[tid] = mx;
  __syncthreads();
  for (int s = 128; s > 0; s >>= 1) {
    if (tid < s) red[tid] = fmaxf(red[tid], red[tid + s]);
    __syncthreads();
  }
  mx = red[0];
  __syncthreads();

  float sum = 0.f;
  for (int i = tid; i < N; i += 256) sum += __expf(row[i] - mx);
  red[tid] = sum;
  __syncthreads();
  for (int s = 128; s > 0; s >>= 1) {
    if (tid < s) red[tid] += red[tid + s];
    __syncthreads();
  }
  const float inv = 1.0f / red[0];

  for (int i = tid; i < N; i += 256)
    orow[i] = (bf16)(__expf(row[i] - mx) * inv);
}

// ---- output: o[c][m] = sum_n v[c][n] attnT[m][n];  out = g*o + x -----------
// 2x2 blocked: 32 (c) x 32 (m) per wave.

__device__ inline void store_out_tile(float* __restrict__ out,
                                      const float* __restrict__ x,
                                      int N, int tc, int tm, int lane,
                                      v8f acc, float g) {
  const int m  = tm + (lane & 15);
  const int c0 = tc + ((lane >> 4) ? 8 : 0);
#pragma unroll
  for (int j = 0; j < 8; ++j) {
    size_t idx = (size_t)(c0 + j) * N + m;
    out[idx] = g * acc[j] + x[idx];
  }
}

__global__ __launch_bounds__(128)
void out_kernel(const bf16* __restrict__ v, const bf16* __restrict__ attnT,
                const float* __restrict__ x, float* __restrict__ out,
                const float* __restrict__ gamma, int C, int N) {
  const int lane = threadIdx.x & 31;
  const int wave = threadIdx.x >> 5;
  const int tile = blockIdx.x * 4 + wave;        // (C/32)*(N/32) super-tiles
  const int tcN  = C >> 5;                       // 8
  const int tc   = (tile % tcN) * 32;
  const int tm   = (tile / tcN) * 32;

  v8f a00 = {}, a01 = {}, a10 = {}, a11 = {};
  for (int nt = 0; nt < N; nt += 32) {
    __builtin_prefetch(attnT + (size_t)tm * N + nt + 64, 0, 0);
    v16bf va0 = load_a_frag(v + (size_t)(tc     ) * N + nt, N, lane);
    v16bf va1 = load_a_frag(v + (size_t)(tc + 16) * N + nt, N, lane);
    v16bf pb0 = load_b_frag(attnT + (size_t)(tm     ) * N + nt, N, lane);
    v16bf pb1 = load_b_frag(attnT + (size_t)(tm + 16) * N + nt, N, lane);
    a00 = wmma_bf16(va0, pb0, a00);
    a01 = wmma_bf16(va0, pb1, a01);
    a10 = wmma_bf16(va1, pb0, a10);
    a11 = wmma_bf16(va1, pb1, a11);
  }

  const float g = gamma[0];
  store_out_tile(out, x, N, tc,      tm,      lane, a00, g);
  store_out_tile(out, x, N, tc,      tm + 16, lane, a01, g);
  store_out_tile(out, x, N, tc + 16, tm,      lane, a10, g);
  store_out_tile(out, x, N, tc + 16, tm + 16, lane, a11, g);
}

// ---------------------------------------------------------------------------

extern "C" void kernel_launch(void* const* d_in, const int* in_sizes, int n_in,
                              void* d_out, int out_size, void* d_ws, size_t ws_size,
                              hipStream_t stream) {
  (void)in_sizes; (void)n_in; (void)out_size; (void)ws_size;
  const int B = 8, C = 256, H = 64, W = 64, N = H * W;

  const float* x     = (const float*)d_in[0];
  const float* Wq    = (const float*)d_in[1];
  const float* bq    = (const float*)d_in[2];
  const float* Wk    = (const float*)d_in[3];
  const float* bk    = (const float*)d_in[4];
  const float* Wv    = (const float*)d_in[5];
  const float* bv    = (const float*)d_in[6];
  const float* gamma = (const float*)d_in[7];
  float* out = (float*)d_out;

  // workspace carve-up (256B aligned)
  char*  ws  = (char*)d_ws;
  size_t off = 0;
  auto carve = [&](size_t bytes) -> void* {
    void* p = ws + off;
    off = (off + bytes + 255) & ~(size_t)255;
    return p;
  };
  bf16*  Wqb   = (bf16*)carve((size_t)C * C * sizeof(bf16));
  bf16*  Wkb   = (bf16*)carve((size_t)C * C * sizeof(bf16));
  bf16*  Wvb   = (bf16*)carve((size_t)C * C * sizeof(bf16));
  bf16*  xT    = (bf16*)carve((size_t)N * C * sizeof(bf16));
  bf16*  qT    = (bf16*)carve((size_t)N * C * sizeof(bf16));
  bf16*  kT    = (bf16*)carve((size_t)N * C * sizeof(bf16));
  bf16*  vbuf  = (bf16*)carve((size_t)C * N * sizeof(bf16));
  float* sT    = (float*)carve((size_t)N * N * sizeof(float));   // 64 MB
  bf16*  attnT = (bf16*)carve((size_t)N * N * sizeof(bf16));     // 32 MB

  // pack weights once
  {
    int n = C * C;
    pack_w_kernel<<<(n + 255) / 256, 256, 0, stream>>>(Wq, Wqb, n);
    pack_w_kernel<<<(n + 255) / 256, 256, 0, stream>>>(Wk, Wkb, n);
    pack_w_kernel<<<(n + 255) / 256, 256, 0, stream>>>(Wv, Wvb, n);
  }

  const int qkvTiles = (C / 16) * (N / 16);        // 4096 (16x16 per wave)
  const int qkTiles  = (N / 32) * (N / 32);        // 16384 (32x32 per wave)
  const int outTiles = (C / 32) * (N / 32);        // 1024  (32x32 per wave)

  for (int b = 0; b < B; ++b) {
    const float* xb = x + (size_t)b * C * N;
    float* ob       = out + (size_t)b * C * N;

    pack_xT_kernel<<<(N * C + 255) / 256, 256, 0, stream>>>(xb, xT, C, N);

    qkv_kernel<<<dim3(qkvTiles / 4, 1, 3), 128, 0, stream>>>(
        Wqb, Wkb, Wvb, bq, bk, bv, xT, qT, kT, vbuf, C, N);

    qk_kernel<<<qkTiles / 4, 128, 0, stream>>>(qT, kT, sT, C, N, 1.0f / W);

    softmax_kernel<<<N, 256, 0, stream>>>(sT, attnT, N);

    out_kernel<<<outTiles / 4, 128, 0, stream>>>(vbuf, attnT, xb, ob, gamma, C, N);
  }
}
